// RNNetwork_85194971284130
// MI455X (gfx1250) — matrix operand order repaired and checked by reading, not compiled, not run on hardware
//
#include <hip/hip_runtime.h>
#include <hip/hip_bf16.h>
#include <stdint.h>

// ---------------------------------------------------------------------------
// Types for CDNA5 WMMA (wave32): v_wmma_f32_16x16x32_bf16
// ---------------------------------------------------------------------------
typedef __attribute__((ext_vector_type(16))) __bf16 v16bf;
typedef __attribute__((ext_vector_type(8)))  __bf16 v8bf;
typedef __attribute__((ext_vector_type(8)))  float  v8f;
typedef __attribute__((ext_vector_type(4)))  float  f32x4;
typedef __attribute__((ext_vector_type(4)))  unsigned int u32x4;
typedef __attribute__((ext_vector_type(8)))  int          i32x8;
typedef __attribute__((ext_vector_type(4)))  int          i32x4;

__device__ __forceinline__ v16bf cat8(v8bf lo, v8bf hi) {
    return __builtin_shufflevector(lo, hi, 0,1,2,3,4,5,6,7,8,9,10,11,12,13,14,15);
}

__device__ __forceinline__ v8f wmma_bf16(v16bf a, v16bf b, v8f c) {
    return __builtin_amdgcn_wmma_f32_16x16x32_bf16(false, a, false, b, (short)0, c,
                                                   false, false);
}

// A-fragment 16x32 bf16; `A` already offset to (m0,k0); stride in elements.
// Lane L: row m0+(L&15), K = koff..+7 and koff+16..+23, koff=(L>>4)*8.
__device__ __forceinline__ v16bf load_a_frag(const __bf16* A, long row_stride, int lane) {
    const __bf16* p = A + (long)(lane & 15) * row_stride + ((lane >> 4) << 3);
    v8bf lo = *(const v8bf*)p;
    v8bf hi = *(const v8bf*)(p + 16);
    return cat8(lo, hi);
}

// B-fragment 32x16 bf16 from Bt[N,K] (row-major); `Bt` offset to (n0,k0).
// Lane L: col n0+(L&15), K = koff..+7 and koff+16..+23, koff=(L>>4)*8.
__device__ __forceinline__ v16bf load_b_frag(const __bf16* Bt, long ldb, int lane) {
    const __bf16* p = Bt + (long)(lane & 15) * ldb + ((lane >> 4) << 3);
    v8bf lo = *(const v8bf*)p;
    v8bf hi = *(const v8bf*)(p + 16);
    return cat8(lo, hi);
}

// ---------------------------------------------------------------------------
// Tensor Data Mover: 2-D tile (rows x row_elems bf16) global -> LDS.
// LDS padding: 4 DWORDs every 256 DWORDs (one 512-bf16 row) -> 1040 B row
// stride, bank-conflict-free fragment reads. Descriptor packing per
// CDNA5 ISA (§8.3/8.4); type=2. 6-arg builtin form.
// ---------------------------------------------------------------------------
#define LDSROW 520   // elements per LDS row (512 data + 8 pad)

__device__ __forceinline__ void tdm_load_2d(unsigned lds_off, const void* gaddr,
                                            unsigned rows, unsigned row_elems,
                                            unsigned long row_stride_elems) {
    unsigned long ga = (unsigned long)(uintptr_t)gaddr;
    u32x4 g0;
    g0.x = 1u;                                               // count=1 (valid user D#)
    g0.y = lds_off;                                          // lds_addr (bytes)
    g0.z = (unsigned)(ga & 0xffffffffu);                     // global_addr[31:0]
    g0.w = (unsigned)((ga >> 32) & 0x1ffffffu) | (2u << 30); // addr[56:32] | type=2
    i32x8 g1;
    g1[0] = (1 << 16)              // data_size = 2 bytes
          | (1 << 20)              // pad_enable
          | (7 << 22)              // pad_interval: 256 DWORDs
          | (3 << 25);             // pad_amount: 4 DWORDs
    g1[1] = (int)((row_elems & 0xffffu) << 16);              // tensor_dim0[15:0]
    g1[2] = (int)((row_elems >> 16) & 0xffffu) | (int)((rows & 0xffffu) << 16);
    g1[3] = (int)((rows >> 16) & 0xffffu) | (int)((row_elems & 0xffffu) << 16); // tile_dim0
    g1[4] = (int)(rows & 0xffffu);                           // tile_dim1 (tile_dim2=0)
    g1[5] = (int)(row_stride_elems & 0xffffffffu);           // tensor_dim0_stride lo
    g1[6] = (int)((row_stride_elems >> 32) & 0xffffu);       // stride hi | dim1_stride=0
    g1[7] = 0;
    i32x4 z4 = {0, 0, 0, 0};
    i32x8 z8 = {0, 0, 0, 0, 0, 0, 0, 0};
    __builtin_amdgcn_tensor_load_to_lds(g0, g1, z4, z4, z8, 0);
}

// ---------------------------------------------------------------------------
// Batched GEMM + bias: C[M,N] f32 = A[M,K] bf16 * Bt[N,K] bf16 + bias.
// 256 thr = 8 waves, tile 128(M) x 64(N), wave = 64x16. K loop unrolled by 2
// with explicit ping/pong fragment sets: no cross-buffer register copies,
// loads for chunk k+1/k+2 in flight while chunk k's 4 WMMAs issue.
// K must be a multiple of 64 (true: 128, 512). grid = (M/128, N/64).
// ---------------------------------------------------------------------------
__global__ void __launch_bounds__(256)
wmma_gemm_bias(const __bf16* __restrict__ A, const __bf16* __restrict__ Bt,
               const float* __restrict__ bias0, const float* __restrict__ bias1,
               float* __restrict__ C, int N, int K) {
    const int lane = threadIdx.x & 31;
    const int wave = threadIdx.x >> 5;
    const int m0 = blockIdx.x * 128 + (wave >> 2) * 64;
    const int n0 = blockIdx.y * 64 + (wave & 3) * 16;

    const __bf16* bp = Bt + (long)n0 * K;
    const __bf16* ap0 = A + (long)m0 * K;
    v8f acc[4] = {};

    // ping buffers: chunk k0
    v16bf b0 = load_b_frag(bp, K, lane);
    v16bf a0[4];
#pragma unroll
    for (int i = 0; i < 4; ++i) a0[i] = load_a_frag(ap0 + (long)i * 16 * K, K, lane);

    int k0 = 0;
    for (; k0 + 64 < K; k0 += 64) {
        // pong: chunk k0+32
        v16bf b1 = load_b_frag(bp + k0 + 32, K, lane);
        v16bf a1[4];
#pragma unroll
        for (int i = 0; i < 4; ++i)
            a1[i] = load_a_frag(ap0 + (long)i * 16 * K + k0 + 32, K, lane);
#pragma unroll
        for (int i = 0; i < 4; ++i) acc[i] = wmma_bf16(a0[i], b0, acc[i]);
        // ping: chunk k0+64
        b0 = load_b_frag(bp + k0 + 64, K, lane);
#pragma unroll
        for (int i = 0; i < 4; ++i)
            a0[i] = load_a_frag(ap0 + (long)i * 16 * K + k0 + 64, K, lane);
#pragma unroll
        for (int i = 0; i < 4; ++i) acc[i] = wmma_bf16(a1[i], b1, acc[i]);
    }
    // tail: chunks k0 (in ping) and k0+32
    {
        v16bf b1 = load_b_frag(bp + k0 + 32, K, lane);
        v16bf a1[4];
#pragma unroll
        for (int i = 0; i < 4; ++i)
            a1[i] = load_a_frag(ap0 + (long)i * 16 * K + k0 + 32, K, lane);
#pragma unroll
        for (int i = 0; i < 4; ++i) acc[i] = wmma_bf16(a0[i], b0, acc[i]);
#pragma unroll
        for (int i = 0; i < 4; ++i) acc[i] = wmma_bf16(a1[i], b1, acc[i]);
    }

    const int col = n0 + (lane & 15);
    float bsum = bias0[col];
    if (bias1) bsum += bias1[col];
    const int rbase = (lane >> 4) << 3;
#pragma unroll
    for (int i = 0; i < 4; ++i) {
#pragma unroll
        for (int r = 0; r < 8; ++r) {
            int row = m0 + i * 16 + rbase + r;
            C[(long)row * N + col] = acc[i][r] + bsum;
        }
    }
}

// ---------------------------------------------------------------------------
// Persistent per-layer recurrence: h_t = relu(ig_t + h_{t-1} @ W_hh).
// grid = 32 WGs x 128 thr (4 waves). WG owns 16 N-cols; waves own 16-batch
// tiles. W_hh strip TDM'd to LDS once, its 16 B-fragments pinned in VGPRs for
// all T steps. Per step: 4 parallel TDM loads stage h_{t-1} (64x512 bf16) in
// LDS, igates prefetched during the DMA, then 16 x (ds_load + WMMA), relu,
// bf16 store, device-scope release/acquire grid barrier.
// Dynamic LDS: (16+64) rows * 1040 B = 83,200 B.
// ---------------------------------------------------------------------------
__global__ void __launch_bounds__(128)
rnn_layer_scan(const __bf16* __restrict__ Whh_t,   // [H,H] pre-transposed (N-major)
               const float*  __restrict__ ig,      // [B,T,H] f32
               __bf16* __restrict__ y,              // [B,T,H] bf16 out
               unsigned* __restrict__ counter,
               int T) {
    const int H = 512;
    extern __shared__ char smem[];
    __bf16* ldsW = (__bf16*)smem;                          // 16 x LDSROW
    __bf16* ldsA = (__bf16*)(smem + 16 * LDSROW * 2);      // 64 x LDSROW
    const unsigned ldsW_off = (unsigned)(uintptr_t)ldsW;
    const unsigned ldsA_off = (unsigned)(uintptr_t)ldsA;

    const int lane = threadIdx.x & 31;
    const int wave = threadIdx.x >> 5;
    const int n0 = blockIdx.x * 16;
    const int m0 = wave * 16;                 // batch tile
    const long TH = (long)T * H;
    const int col = n0 + (lane & 15);
    const int rbase = m0 + ((lane >> 4) << 3);

    // Stage this WG's W_hh strip (16 x 512) once via TDM.
    if (wave == 0)
        tdm_load_2d(ldsW_off, Whh_t + (long)n0 * H, 16, H, H);
    __builtin_amdgcn_s_wait_tensorcnt(0);
    __syncthreads();

    // Pin all 16 B-fragments (full K) in VGPRs for the whole scan.
    v16bf bfr[16];
#pragma unroll
    for (int c = 0; c < 16; ++c)
        bfr[c] = load_b_frag(ldsW + c * 32, LDSROW, lane);

    for (int t = 0; t < T; ++t) {
        if (t > 0) {
            // Each wave DMAs its 16 rows of h_{t-1} into LDS.
            tdm_load_2d(ldsA_off + (unsigned)(wave * 16) * (LDSROW * 2),
                        y + (long)(wave * 16) * TH + (long)(t - 1) * H,
                        16, H, (unsigned long)TH);
        }
        // Prefetch igates (independent of the DMA).
        const long obase = (long)rbase * TH + (long)t * H + col;
        float igv[8];
#pragma unroll
        for (int r = 0; r < 8; ++r) igv[r] = ig[obase + (long)r * TH];

        __builtin_amdgcn_s_wait_tensorcnt(0);
        __syncthreads();

        v8f acc = {};
        if (t > 0) {
#pragma unroll
            for (int c = 0; c < 16; ++c) {
                v16bf af = load_a_frag(ldsA + m0 * LDSROW + c * 32, LDSROW, lane);
                acc = wmma_bf16(af, bfr[c], acc);
            }
        }
#pragma unroll
        for (int r = 0; r < 8; ++r) {
            float v = acc[r] + igv[r];
            y[obase + (long)r * TH] = (__bf16)(v > 0.f ? v : 0.f);
        }

        // ---- grid-wide barrier between timesteps ----
        __threadfence();
        __syncthreads();
        if (threadIdx.x == 0) {
            __hip_atomic_fetch_add(counter, 1u, __ATOMIC_RELEASE,
                                   __HIP_MEMORY_SCOPE_AGENT);
            const unsigned target = (unsigned)(t + 1) * gridDim.x;
            while (__hip_atomic_load(counter, __ATOMIC_ACQUIRE,
                                     __HIP_MEMORY_SCOPE_AGENT) < target) {
                __builtin_amdgcn_s_sleep(2);
            }
        }
        __syncthreads();
    }
}

// ---------------------------------------------------------------------------
// Helpers: vectorized f32->bf16 convert (8 elems/thread) and
// f32 [K,N] -> bf16 [N,K] transpose.
// ---------------------------------------------------------------------------
__global__ void __launch_bounds__(256)
cvt_to_bf16_v8(const float* __restrict__ in, __bf16* __restrict__ out, int n8) {
    int i = blockIdx.x * 256 + threadIdx.x;
    if (i >= n8) return;
    const f32x4* p = (const f32x4*)(in + (size_t)i * 8);
    f32x4 a = p[0], b = p[1];
    v8bf o;
    o[0] = (__bf16)a.x; o[1] = (__bf16)a.y; o[2] = (__bf16)a.z; o[3] = (__bf16)a.w;
    o[4] = (__bf16)b.x; o[5] = (__bf16)b.y; o[6] = (__bf16)b.z; o[7] = (__bf16)b.w;
    *(v8bf*)(out + (size_t)i * 8) = o;
}

__global__ void __launch_bounds__(256)
transpose_to_bf16(const float* __restrict__ W, __bf16* __restrict__ Wt, int K, int N) {
    int k = blockIdx.x * 16 + (threadIdx.x & 15);
    int n = blockIdx.y * 16 + (threadIdx.x >> 4);
    Wt[(long)n * K + k] = (__bf16)W[(long)k * N + n];
}

// ---------------------------------------------------------------------------
extern "C" void kernel_launch(void* const* d_in, const int* in_sizes, int n_in,
                              void* d_out, int out_size, void* d_ws, size_t ws_size,
                              hipStream_t stream) {
    (void)in_sizes; (void)n_in; (void)out_size; (void)ws_size;
    const int B = 64, T = 512, I = 128, H = 512;
    const int M = B * T;  // 32768 tokens

    const float* x = (const float*)d_in[0];
    const float* W_ih[3] = {(const float*)d_in[1], (const float*)d_in[5], (const float*)d_in[9]};
    const float* W_hh[3] = {(const float*)d_in[2], (const float*)d_in[6], (const float*)d_in[10]};
    const float* b_ih[3] = {(const float*)d_in[3], (const float*)d_in[7], (const float*)d_in[11]};
    const float* b_hh[3] = {(const float*)d_in[4], (const float*)d_in[8], (const float*)d_in[12]};
    const float* W_out = (const float*)d_in[13];
    const float* b_out = (const float*)d_in[14];
    float* out = (float*)d_out;

    // ---- workspace carve-up (deterministic, ~139 MB) ----
    char* ws = (char*)d_ws;
    size_t off = 0;
    auto alloc = [&](size_t bytes) -> void* {
        void* p = ws + off;
        off += (bytes + 255) & ~(size_t)255;
        return p;
    };
    unsigned* counters = (unsigned*)alloc(256);
    __bf16* wiht[3], *whht[3];
    wiht[0] = (__bf16*)alloc((size_t)I * H * 2);
    whht[0] = (__bf16*)alloc((size_t)H * H * 2);
    wiht[1] = (__bf16*)alloc((size_t)H * H * 2);
    whht[1] = (__bf16*)alloc((size_t)H * H * 2);
    wiht[2] = (__bf16*)alloc((size_t)H * H * 2);
    whht[2] = (__bf16*)alloc((size_t)H * H * 2);
    __bf16* woutt = (__bf16*)alloc((size_t)H * I * 2);
    __bf16* xbf   = (__bf16*)alloc((size_t)M * I * 2);      // 8 MB
    __bf16* yA    = (__bf16*)alloc((size_t)M * H * 2);      // 32 MB
    __bf16* yB    = (__bf16*)alloc((size_t)M * H * 2);      // 32 MB
    float*  igbuf = (float*) alloc((size_t)M * H * 4);      // 64 MB

    (void)hipMemsetAsync(counters, 0, 256, stream);

    cvt_to_bf16_v8<<<(M * I / 8 + 255) / 256, 256, 0, stream>>>(x, xbf, M * I / 8);

    transpose_to_bf16<<<dim3(I / 16, H / 16), 256, 0, stream>>>(W_ih[0], wiht[0], I, H);
    transpose_to_bf16<<<dim3(H / 16, H / 16), 256, 0, stream>>>(W_hh[0], whht[0], H, H);
    transpose_to_bf16<<<dim3(H / 16, H / 16), 256, 0, stream>>>(W_ih[1], wiht[1], H, H);
    transpose_to_bf16<<<dim3(H / 16, H / 16), 256, 0, stream>>>(W_hh[1], whht[1], H, H);
    transpose_to_bf16<<<dim3(H / 16, H / 16), 256, 0, stream>>>(W_ih[2], wiht[2], H, H);
    transpose_to_bf16<<<dim3(H / 16, H / 16), 256, 0, stream>>>(W_hh[2], whht[2], H, H);
    transpose_to_bf16<<<dim3(H / 16, I / 16), 256, 0, stream>>>(W_out, woutt, H, I);

    const dim3 gemmBlk(256);
    const dim3 scanGrid(H / 16), scanBlk(128);
    const size_t scanLds = (size_t)(16 + 64) * LDSROW * 2;   // 83,200 B

    wmma_gemm_bias<<<dim3(M / 128, H / 64), gemmBlk, 0, stream>>>(
        xbf, wiht[0], b_ih[0], b_hh[0], igbuf, H, I);
    rnn_layer_scan<<<scanGrid, scanBlk, scanLds, stream>>>(whht[0], igbuf, yA, counters + 0, T);

    wmma_gemm_bias<<<dim3(M / 128, H / 64), gemmBlk, 0, stream>>>(
        yA, wiht[1], b_ih[1], b_hh[1], igbuf, H, H);
    rnn_layer_scan<<<scanGrid, scanBlk, scanLds, stream>>>(whht[1], igbuf, yB, counters + 1, T);

    wmma_gemm_bias<<<dim3(M / 128, H / 64), gemmBlk, 0, stream>>>(
        yB, wiht[2], b_ih[2], b_hh[2], igbuf, H, H);
    rnn_layer_scan<<<scanGrid, scanBlk, scanLds, stream>>>(whht[2], igbuf, yA, counters + 2, T);

    wmma_gemm_bias<<<dim3(M / 128, I / 64), gemmBlk, 0, stream>>>(
        yA, woutt, b_out, (const float*)nullptr, out, I, H);
}
